// SpatialAttention_21268678050427
// MI455X (gfx1250) — matrix operand order repaired and checked by reading, not compiled, not run on hardware
//
#include <hip/hip_runtime.h>
#include <hip/hip_bf16.h>
#include <stdint.h>

typedef __attribute__((ext_vector_type(16))) _Float16 v16h;
typedef __attribute__((ext_vector_type(8)))  _Float16 v8h;
typedef __attribute__((ext_vector_type(8)))  float    v8f;

#define BATCH 4
#define CH    256
#define HH    64
#define WW    64
#define NN    (HH*WW)   // 4096
#define DQK   32
#define KB    64        // keys processed per mainloop iteration
#define KSTR  40        // lK row stride in halves (80B rows: 16B-aligned, conflict-free)
#define VSTR  72        // lV/lP row stride in halves (144B rows: 16B-aligned, conflict-free)

// ---- helpers --------------------------------------------------------------
__device__ __forceinline__ v16h ld_bfrag(const _Float16* p) {   // 16 contiguous halves
  v8h a0 = *(const v8h*)p;
  v8h a1 = *(const v8h*)(p + 8);
  v16h r;
  #pragma unroll
  for (int i = 0; i < 8; ++i) { r[i] = a0[i]; r[8 + i] = a1[i]; }
  return r;
}
__device__ __forceinline__ v16h ld_afrag(const _Float16* p) {   // halves p[0..7], p[16..23]
  v8h a0 = *(const v8h*)p;
  v8h a1 = *(const v8h*)(p + 16);
  v16h r;
  #pragma unroll
  for (int i = 0; i < 8; ++i) { r[i] = a0[i]; r[8 + i] = a1[i]; }
  return r;
}
// CDNA5 async global->LDS copy, 16B per lane (tracked by ASYNCcnt).
__device__ __forceinline__ void async_b128(unsigned lds_off, const _Float16* g) {
  asm volatile("global_load_async_to_lds_b128 %0, %1, off"
               :: "v"(lds_off), "v"((uint64_t)(uintptr_t)g)
               : "memory");
}
__device__ __forceinline__ void wait_async0() {
  asm volatile("s_wait_asynccnt 0x0" ::: "memory");
}

// ---------------------------------------------------------------------------
// proj q (1x3, pad W) / k (3x1, pad H). Thread = (b,n) x 8-channel group
// (gridDim.y = 4). Weight reads are lane-uniform -> scalar loads.
// Outputs f16 [b][n][32].
// ---------------------------------------------------------------------------
__global__ __launch_bounds__(256) void proj_qk_kernel(
    const float* __restrict__ x,
    const float* __restrict__ Wq, const float* __restrict__ bq,
    const float* __restrict__ Wk, const float* __restrict__ bk,
    _Float16* __restrict__ Qh, _Float16* __restrict__ Kh)
{
  const int id = blockIdx.x * blockDim.x + threadIdx.x;   // b*NN + n
  const int o0 = blockIdx.y * 8;
  const int b = id >> 12;
  const int n = id & (NN - 1);
  const int h = n >> 6, w = n & (WW - 1);

  float qa[8], ka[8];
  #pragma unroll
  for (int o = 0; o < 8; ++o) { qa[o] = 0.f; ka[o] = 0.f; }

  const float* xb = x + (size_t)b * CH * NN;
  #pragma unroll 1
  for (int c = 0; c < CH; ++c) {
    const float* xc = xb + (size_t)c * NN;
    const float xm = xc[n];
    const float xl = (w > 0)      ? xc[n - 1]  : 0.f;
    const float xr = (w < WW - 1) ? xc[n + 1]  : 0.f;
    const float xu = (h > 0)      ? xc[n - WW] : 0.f;
    const float xd = (h < HH - 1) ? xc[n + WW] : 0.f;
    #pragma unroll
    for (int o = 0; o < 8; ++o) {
      const float* wq = Wq + ((size_t)(o0 + o) * CH + c) * 3;
      const float* wk = Wk + ((size_t)(o0 + o) * CH + c) * 3;
      qa[o] += xl * wq[0] + xm * wq[1] + xr * wq[2];
      ka[o] += xu * wk[0] + xm * wk[1] + xd * wk[2];
    }
  }
  _Float16* qo = Qh + (size_t)id * DQK + o0;
  _Float16* ko = Kh + (size_t)id * DQK + o0;
  #pragma unroll
  for (int o = 0; o < 8; ++o) {
    qo[o] = (_Float16)(qa[o] + bq[o0 + o]);
    ko[o] = (_Float16)(ka[o] + bk[o0 + o]);
  }
}

// ---------------------------------------------------------------------------
// proj v (1x1 conv): thread per (b, n, 16-channel group). Output Vt[b][c][n].
// ---------------------------------------------------------------------------
__global__ __launch_bounds__(256) void proj_v_kernel(
    const float* __restrict__ x, const float* __restrict__ Wv,
    const float* __restrict__ bv, _Float16* __restrict__ Vt)
{
  const int idx = blockIdx.x * blockDim.x + threadIdx.x;
  const int n  = idx & (NN - 1);
  const int r  = idx >> 12;
  const int b  = r & 3;
  const int c0 = (r >> 2) * 16;

  float acc[16];
  #pragma unroll
  for (int j = 0; j < 16; ++j) acc[j] = 0.f;

  const float* xb = x + (size_t)b * CH * NN;
  #pragma unroll 1
  for (int ci = 0; ci < CH; ++ci) {
    const float xv = xb[(size_t)ci * NN + n];
    #pragma unroll
    for (int j = 0; j < 16; ++j)
      acc[j] += xv * Wv[(size_t)(c0 + j) * CH + ci];
  }
  #pragma unroll
  for (int j = 0; j < 16; ++j)
    Vt[((size_t)b * CH + c0 + j) * NN + n] = (_Float16)(acc[j] + bv[c0 + j]);
}

// ---------------------------------------------------------------------------
// Fused flash attention, 64-key blocks, double-buffered async LDS staging.
// Block = 4 waves = 32 query rows. Wave (qt, chalf): query tile iq0+qt*16,
// channels chalf*128..+127. Per iteration: issue next block's async DMA,
// 4 WMMAs for S, online max (shuffle) with deferred per-lane row-sums,
// P->LDS, 16 WMMAs for PV, then wait-async + single barrier.
// ---------------------------------------------------------------------------
__global__ __launch_bounds__(128) void attn_kernel(
    const _Float16* __restrict__ Qh, const _Float16* __restrict__ Kh,
    const _Float16* __restrict__ Vt, float* __restrict__ out)
{
  __shared__ _Float16 lK[2][KB * KSTR];     // [key][d]     2 x 5 KB
  __shared__ _Float16 lV[2][CH * VSTR];     // [chan][key]  2 x 36 KB
  __shared__ _Float16 lP[4][16 * VSTR];     // per-wave P   9 KB

  const int b    = blockIdx.y;
  const int iq0  = blockIdx.x * 32;
  const int tid  = threadIdx.x;
  const int wv   = tid >> 5;
  const int lane = tid & 31;
  const int qt   = wv >> 1;
  const int cb   = (wv & 1) * 128;
  const int l15  = lane & 15;
  const int hi   = lane >> 4;
  const int ak0  = hi ? 8 : 0;

  const _Float16* Kb = Kh + (size_t)b * NN * DQK;
  const _Float16* Vb = Vt + (size_t)b * CH * NN;

  // async-stage one 64-key block (K: 256 chunks, V: 2048 chunks of 16B)
  auto stage = [&](int j0, int p) {
    const unsigned kbase = (unsigned)(uintptr_t)&lK[p][0];
    const unsigned vbase = (unsigned)(uintptr_t)&lV[p][0];
    const int krow = tid >> 1, kc0 = (tid & 1) * 2;
    #pragma unroll
    for (int cc = 0; cc < 2; ++cc) {
      const int chk = kc0 + cc;
      async_b128(kbase + (unsigned)(krow * KSTR + chk * 8) * 2,
                 Kb + (size_t)(j0 + krow) * DQK + chk * 8);
    }
    #pragma unroll
    for (int rr = 0; rr < 2; ++rr) {
      const int vr = tid * 2 + rr;          // channel row 0..255
      #pragma unroll
      for (int chk = 0; chk < 8; ++chk)
        async_b128(vbase + (unsigned)(vr * VSTR + chk * 8) * 2,
                   Vb + (size_t)vr * NN + j0 + chk * 8);
    }
  };

  // ---- Q A-fragment (held for whole kernel) ----
  v16h qa;
  {
    const int qi = iq0 + qt * 16 + l15;
    qa = ld_afrag(Qh + ((size_t)b * NN + qi) * DQK + ak0);
  }

  v8f acc[8];
  #pragma unroll
  for (int t = 0; t < 8; ++t)
    #pragma unroll
    for (int rr = 0; rr < 8; ++rr) acc[t][rr] = 0.f;

  float m_i[8], l_i[8];
  #pragma unroll
  for (int rr = 0; rr < 8; ++rr) { m_i[rr] = -1e30f; l_i[rr] = 0.f; }

  // prologue: stage block 0 into buffer 0
  stage(0, 0);
  wait_async0();
  __syncthreads();

  const int NIT = NN / KB;                  // 64 iterations
  for (int it = 0; it < NIT; ++it) {
    const int p = it & 1;
    if (it + 1 < NIT) stage((it + 1) * KB, p ^ 1);   // overlap DMA with compute

    // ---- S = Q * K^T, four 16-key tiles ----
    v8f s[4];
    #pragma unroll
    for (int jt = 0; jt < 4; ++jt) {
      v16h kb = ld_afrag(&lK[p][(jt * 16 + l15) * KSTR + hi * 8]);
      v8f zero;
      #pragma unroll
      for (int rr = 0; rr < 8; ++rr) zero[rr] = 0.f;
      s[jt] = __builtin_amdgcn_wmma_f32_16x16x32_f16(false, qa, false, kb,
                                                     (short)0, zero, false, false);
    }

    // ---- online softmax: broadcast row-max; row-sum kept per-lane ----
    #pragma unroll
    for (int rr = 0; rr < 8; ++rr) {
      float mb = fmaxf(fmaxf(s[0][rr], s[1][rr]), fmaxf(s[2][rr], s[3][rr]));
      #pragma unroll
      for (int ms = 1; ms < 16; ms <<= 1) mb = fmaxf(mb, __shfl_xor(mb, ms));
      const float mn = fmaxf(m_i[rr], mb);
      const float sc = __expf(m_i[rr] - mn);
      m_i[rr] = mn;
      float psum = 0.f;
      const int prow = (rr + 8 * hi) * VSTR + l15;
      #pragma unroll
      for (int jt = 0; jt < 4; ++jt) {
        const float e = __expf(s[jt][rr] - mn);
        psum += e;
        lP[wv][prow + jt * 16] = (_Float16)e;
      }
      l_i[rr] = l_i[rr] * sc + psum;        // per-lane partial; reduced in epilogue
      #pragma unroll
      for (int t = 0; t < 8; ++t) acc[t][rr] *= sc;
    }

    // ---- P A-fragments (wave-internal LDS round trip) ----
    v16h pa0 = ld_afrag(&lP[wv][l15 * VSTR + ak0]);
    v16h pa1 = ld_afrag(&lP[wv][l15 * VSTR + 32 + ak0]);

    // ---- O += P * V^T over 8 channel tiles x 2 K-chunks ----
    #pragma unroll
    for (int t = 0; t < 8; ++t) {
      v16h vb0 = ld_afrag(&lV[p][(cb + t * 16 + l15) * VSTR + hi * 8]);
      acc[t] = __builtin_amdgcn_wmma_f32_16x16x32_f16(false, pa0, false, vb0,
                                                      (short)0, acc[t], false, false);
    }
    #pragma unroll
    for (int t = 0; t < 8; ++t) {
      v16h vb1 = ld_afrag(&lV[p][(cb + t * 16 + l15) * VSTR + 32 + hi * 8]);
      acc[t] = __builtin_amdgcn_wmma_f32_16x16x32_f16(false, pa1, false, vb1,
                                                      (short)0, acc[t], false, false);
    }

    wait_async0();        // next buffer resident (issued before compute)
    __syncthreads();      // publish next buffer / retire this one
  }

  // ---- epilogue: reduce deferred row-sums, normalize, store out[b][c][i] ----
  float inv[8];
  #pragma unroll
  for (int rr = 0; rr < 8; ++rr) {
    float sum = l_i[rr];
    #pragma unroll
    for (int ms = 1; ms < 16; ms <<= 1) sum += __shfl_xor(sum, ms);
    inv[rr] = 1.f / sum;
  }
  const int ib = iq0 + qt * 16 + 8 * hi;
  #pragma unroll
  for (int t = 0; t < 8; ++t) {
    const int c = cb + t * 16 + l15;
    float* orow = out + ((size_t)b * CH + c) * NN;
    #pragma unroll
    for (int rr = 0; rr < 8; ++rr)
      orow[ib + rr] = acc[t][rr] * inv[rr];
  }
}

// ---------------------------------------------------------------------------
extern "C" void kernel_launch(void* const* d_in, const int* in_sizes, int n_in,
                              void* d_out, int out_size, void* d_ws, size_t ws_size,
                              hipStream_t stream) {
  (void)in_sizes; (void)n_in; (void)out_size; (void)ws_size;
  const float* x  = (const float*)d_in[0];
  const float* Wq = (const float*)d_in[1];
  const float* bq = (const float*)d_in[2];
  const float* Wk = (const float*)d_in[3];
  const float* bk = (const float*)d_in[4];
  const float* Wv = (const float*)d_in[5];
  const float* bv = (const float*)d_in[6];
  float* out = (float*)d_out;

  char* ws = (char*)d_ws;
  const size_t qk_bytes = (size_t)BATCH * NN * DQK * sizeof(_Float16); // 1 MB
  _Float16* Qh = (_Float16*)(ws);
  _Float16* Kh = (_Float16*)(ws + qk_bytes);
  _Float16* Vt = (_Float16*)(ws + 2 * qk_bytes);                       // 8 MB

  proj_qk_kernel<<<dim3((BATCH * NN) / 256, 4), 256, 0, stream>>>(
      x, Wq, bq, Wk, bk, Qh, Kh);
  proj_v_kernel<<<(BATCH * NN * (CH / 16)) / 256, 256, 0, stream>>>(x, Wv, bv, Vt);
  attn_kernel<<<dim3(NN / 32, BATCH), 128, 0, stream>>>(Qh, Kh, Vt, out);
}
